// GQAAttention_29661044146724
// MI455X (gfx1250) — compile-verified, matrix operand order checked by reading, not run on hardware
//
#include <hip/hip_runtime.h>
#include <hip/hip_bf16.h>
#include <stdint.h>

// ---------------------------------------------------------------------------
// GQA + sliding-window attention for MI455X (gfx1250, wave32, WMMA bf16)
// ---------------------------------------------------------------------------

typedef __attribute__((ext_vector_type(16))) __bf16       v16bf;
typedef __attribute__((ext_vector_type(8)))  float        v8f;
typedef __attribute__((ext_vector_type(4)))  unsigned int u32x4;

#define D_MODEL 4096
#define N_Q     32
#define N_KV    8
#define HD      128
#define SLIDE   512
#define SEQ     2048
#define BATCH   2
#define TOKENS  (BATCH * SEQ)   // 4096
#define QF      (N_Q * HD)      // 4096
#define KVF     (N_KV * HD)     // 1024
#define GRP     (N_Q / N_KV)    // 4

// float -> bf16 bits, round-to-nearest-even
__device__ __forceinline__ unsigned short f2bf(float f) {
    unsigned int u = __float_as_uint(f);
    u += 0x7FFFu + ((u >> 16) & 1u);
    return (unsigned short)(u >> 16);
}

// Load a 16x32 (MxK) bf16 A-fragment (or 32x16 KxN B-fragment) for wave32 WMMA.
// `rowk` points at element k0 of THIS lane's row (row = lane&15 chosen by caller).
// Lane group (lane>>4) selects kb = 0 or 8; halves 0..7 = K kb..kb+7,
// halves 8..15 = K kb+16..kb+23  (CDNA5 ISA 7.12.2 16-bit layout).
__device__ __forceinline__ v16bf load_frag(const unsigned short* rowk, int lane) {
    int kb = (lane >> 4) << 3;
    union { v16bf v; u32x4 u[2]; } f;
    f.u[0] = *(const u32x4*)(rowk + kb);
    f.u[1] = *(const u32x4*)(rowk + kb + 16);
    return f.v;
}

__device__ __forceinline__ v8f wmma_bf16(v16bf a, v16bf b, v8f c) {
    // (neg_a, A, neg_b, B, c_mod, C, reuse_a, reuse_b)
    return __builtin_amdgcn_wmma_f32_16x16x32_bf16(false, a, false, b, (short)0, c,
                                                   false, false);
}

// ---------------------------------------------------------------------------
// 1) fp32 -> bf16 conversion
// ---------------------------------------------------------------------------
__global__ __launch_bounds__(256) void cvt_f32_bf16(const float* __restrict__ src,
                                                    unsigned short* __restrict__ dst,
                                                    int n) {
    int i = blockIdx.x * blockDim.x + threadIdx.x;
    if (i < n) dst[i] = f2bf(src[i]);
}

// ---------------------------------------------------------------------------
// 2) NT GEMM: C[M,N] = A[M,K] * B[N,K]^T  (both bf16, row-major along K)
//    mode 0: store bf16 row-major [M,N]
//    mode 1: store f32  row-major [M,N]
//    mode 2: store bf16 V-transposed: [(b*N_KV+h)*HD + d][SEQ] key-major
//    Block = 128 threads = 4 waves (2x2); wave tile 64x64 (16 accumulators),
//    block tile 128x128. Per 32-deep k-step: 16 b128 loads feed 16 WMMAs
//    (A frags reused across 4 cols, B frags across 4 rows -> 4x the
//    arithmetic intensity of a 16x16 wave tile).
// ---------------------------------------------------------------------------
__global__ __launch_bounds__(128) void gemm_nt_bf16(const unsigned short* __restrict__ A,
                                                    const unsigned short* __restrict__ B,
                                                    void* __restrict__ Cout,
                                                    int M, int N, int K, int mode) {
    const int lane = threadIdx.x & 31;
    const int wave = threadIdx.x >> 5;       // 0..3
    const int wm   = wave & 1;               // 2 wave-rows
    const int wn   = wave >> 1;              // 2 wave-cols
    const int m0   = blockIdx.x * 128 + wm * 64;
    const int n0   = blockIdx.y * 128 + wn * 64;

    const unsigned short* ap[4];
    const unsigned short* bp[4];
#pragma unroll
    for (int r = 0; r < 4; ++r) {
        ap[r] = A + (size_t)(m0 + r * 16 + (lane & 15)) * K;
        bp[r] = B + (size_t)(n0 + r * 16 + (lane & 15)) * K;
    }

    v8f acc[4][4];
#pragma unroll
    for (int r = 0; r < 4; ++r)
#pragma unroll
        for (int c = 0; c < 4; ++c)
#pragma unroll
            for (int i = 0; i < 8; ++i) acc[r][c][i] = 0.0f;

    for (int k0 = 0; k0 < K; k0 += 32) {
        v16bf af[4], bf_[4];
#pragma unroll
        for (int r = 0; r < 4; ++r) af[r]  = load_frag(ap[r] + k0, lane);
#pragma unroll
        for (int c = 0; c < 4; ++c) bf_[c] = load_frag(bp[c] + k0, lane);
#pragma unroll
        for (int r = 0; r < 4; ++r)
#pragma unroll
            for (int c = 0; c < 4; ++c)
                acc[r][c] = wmma_bf16(af[r], bf_[c], acc[r][c]);
    }

    const int g   = lane >> 4;
    const int col = lane & 15;
#pragma unroll
    for (int r = 0; r < 4; ++r)
#pragma unroll
        for (int c = 0; c < 4; ++c) {
            const int rowb = m0 + r * 16;
            const int colb = n0 + c * 16 + col;
            if (mode == 1) {
                float* C = (float*)Cout;
#pragma unroll
                for (int i = 0; i < 8; ++i) {
                    int row = rowb + i + 8 * g;
                    C[(size_t)row * N + colb] = acc[r][c][i];
                }
            } else if (mode == 0) {
                unsigned short* C = (unsigned short*)Cout;
#pragma unroll
                for (int i = 0; i < 8; ++i) {
                    int row = rowb + i + 8 * g;
                    C[(size_t)row * N + colb] = f2bf(acc[r][c][i]);
                }
            } else { // mode 2: V transpose store  vt[(b*8+h)*128+d][key]
                unsigned short* C = (unsigned short*)Cout;
#pragma unroll
                for (int i = 0; i < 8; ++i) {
                    int row = rowb + i + 8 * g;      // token index 0..4095
                    int b_  = row >> 11;             // /2048
                    int tok = row & (SEQ - 1);
                    int h   = colb >> 7;             // /128
                    int d   = colb & (HD - 1);
                    C[(((size_t)(b_ * N_KV + h) * HD + d) << 11) + tok] =
                        f2bf(acc[r][c][i]);
                }
            }
        }
}

// ---------------------------------------------------------------------------
// 3) Flash attention, GQA + causal + sliding window (512).
//    grid = (SEQ/16, N_Q, BATCH); block = 1 wave (32 threads).
//    Q,K bf16 row-major [tokens, feat]; V bf16 transposed [(b,h,d), key].
//    Output Ob bf16 [tokens, N_Q*HD].
// ---------------------------------------------------------------------------
__global__ __launch_bounds__(32) void attn_gqa_swa(const unsigned short* __restrict__ Qb,
                                                   const unsigned short* __restrict__ Kb,
                                                   const unsigned short* __restrict__ Vt,
                                                   unsigned short* __restrict__ Ob) {
    __shared__ unsigned short pbuf[16 * 32];   // P tile staged for layout swizzle

    const int lane = threadIdx.x & 31;
    const int qt   = blockIdx.x;               // 16-query tile
    const int hq   = blockIdx.y;               // query head
    const int b    = blockIdx.z;               // batch
    const int hkv  = hq >> 2;                  // GRP = 4
    const int qi0  = qt * 16;
    const int g    = lane >> 4;
    const int col  = lane & 15;

    const float scale   = 0.08838834764831845f;  // 1/sqrt(128)
    const float NEG_INF = -__builtin_inff();

    // Q fragments: 16 rows x 128 dims = 4 fragments of 16x32
    const unsigned short* qrow = Qb + (size_t)(b * SEQ + qi0 + col) * QF + hq * HD;
    v16bf qf[4];
#pragma unroll
    for (int t = 0; t < 4; ++t) qf[t] = load_frag(qrow + t * 32, lane);

    float m8[8], l8[8];
    v8f   o[8];
#pragma unroll
    for (int i = 0; i < 8; ++i) { m8[i] = -1e30f; l8[i] = 0.0f; }
#pragma unroll
    for (int cc = 0; cc < 8; ++cc)
#pragma unroll
        for (int i = 0; i < 8; ++i) o[cc][i] = 0.0f;

    int lo = qi0 - (SLIDE - 1);
    if (lo < 0) lo = 0;
    lo &= ~31;                                 // align chunk start
    const int hi = qi0 + 15;

    for (int kc = lo; kc <= hi; kc += 32) {
        // ---- S = Q K^T for keys [kc, kc+32) : two 16x16 f32 tiles ----
        v8f s0, s1;
#pragma unroll
        for (int i = 0; i < 8; ++i) { s0[i] = 0.0f; s1[i] = 0.0f; }

        const unsigned short* krow0 = Kb + (size_t)(b * SEQ + kc + col) * KVF + hkv * HD;
        const unsigned short* krow1 = krow0 + (size_t)16 * KVF;
#pragma unroll
        for (int t = 0; t < 4; ++t) {
            v16bf kf0 = load_frag(krow0 + t * 32, lane);
            s0 = wmma_bf16(qf[t], kf0, s0);
            v16bf kf1 = load_frag(krow1 + t * 32, lane);
            s1 = wmma_bf16(qf[t], kf1, s1);
        }

        // ---- scale + causal/sliding-window mask + row max ----
        float rm[8];
#pragma unroll
        for (int i = 0; i < 8; ++i) {
            int r  = qi0 + i + 8 * g;
            int j0 = kc + col;
            int j1 = kc + 16 + col;
            float x0 = s0[i] * scale;
            float x1 = s1[i] * scale;
            if (!(j0 <= r && (r - j0) < SLIDE)) x0 = NEG_INF;
            if (!(j1 <= r && (r - j1) < SLIDE)) x1 = NEG_INF;
            s0[i] = x0; s1[i] = x1;
            rm[i] = fmaxf(x0, x1);
        }
#pragma unroll
        for (int msk = 1; msk <= 8; msk <<= 1)
#pragma unroll
            for (int i = 0; i < 8; ++i) rm[i] = fmaxf(rm[i], __shfl_xor(rm[i], msk, 32));

        // ---- online softmax update ----
        float p0[8], p1[8], rs[8], alpha[8];
#pragma unroll
        for (int i = 0; i < 8; ++i) {
            float mn = fmaxf(m8[i], rm[i]);
            alpha[i] = __expf(m8[i] - mn);
            m8[i]    = mn;
            p0[i]    = __expf(s0[i] - mn);
            p1[i]    = __expf(s1[i] - mn);
            rs[i]    = p0[i] + p1[i];
        }
#pragma unroll
        for (int msk = 1; msk <= 8; msk <<= 1)
#pragma unroll
            for (int i = 0; i < 8; ++i) rs[i] += __shfl_xor(rs[i], msk, 32);
#pragma unroll
        for (int i = 0; i < 8; ++i) l8[i] = l8[i] * alpha[i] + rs[i];
#pragma unroll
        for (int cc = 0; cc < 8; ++cc)
#pragma unroll
            for (int i = 0; i < 8; ++i) o[cc][i] *= alpha[i];

        // ---- stage P (C-layout) into LDS, reload in A-fragment layout ----
#pragma unroll
        for (int i = 0; i < 8; ++i) {
            int r = i + 8 * g;
            pbuf[r * 32 + col]      = f2bf(p0[i]);
            pbuf[r * 32 + 16 + col] = f2bf(p1[i]);
        }
        __syncthreads();
        v16bf pf = load_frag(pbuf + col * 32, lane);   // A-frag row = lane&15

        // ---- O += P * V : 8 output chunks of 16 dims ----
#pragma unroll
        for (int cc = 0; cc < 8; ++cc) {
            const unsigned short* vrow =
                Vt + ((size_t)((b * N_KV + hkv) * HD) + cc * 16 + col) * SEQ + kc;
            v16bf vf = load_frag(vrow, lane);          // B-frag: k = key, n = dim
            o[cc] = wmma_bf16(pf, vf, o[cc]);
        }
        __syncthreads();
    }

    // ---- normalize and store bf16 output [tokens, N_Q*HD] ----
    float inv[8];
#pragma unroll
    for (int i = 0; i < 8; ++i) inv[i] = 1.0f / l8[i];
#pragma unroll
    for (int cc = 0; cc < 8; ++cc)
#pragma unroll
        for (int i = 0; i < 8; ++i) {
            int row = b * SEQ + qi0 + i + 8 * g;
            Ob[(size_t)row * QF + hq * HD + cc * 16 + col] = f2bf(o[cc][i] * inv[i]);
        }
}

// ---------------------------------------------------------------------------
// Host-side orchestration
// ---------------------------------------------------------------------------
extern "C" void kernel_launch(void* const* d_in, const int* in_sizes, int n_in,
                              void* d_out, int out_size, void* d_ws, size_t ws_size,
                              hipStream_t stream) {
    const float* x  = (const float*)d_in[0];   // [2,2048,4096]
    const float* wq = (const float*)d_in[1];   // [4096,4096]
    const float* wk = (const float*)d_in[2];   // [1024,4096]
    const float* wv = (const float*)d_in[3];   // [1024,4096]
    const float* wo = (const float*)d_in[4];   // [4096,4096]
    float* out = (float*)d_out;                // [2,2048,4096]

    unsigned short* ws = (unsigned short*)d_ws;
    size_t off = 0;
    const size_t nX  = (size_t)TOKENS * D_MODEL;  // 16.78M
    const size_t nWq = (size_t)QF * D_MODEL;
    const size_t nWk = (size_t)KVF * D_MODEL;

    unsigned short* xb  = ws + off; off += nX;
    unsigned short* wqb = ws + off; off += nWq;
    unsigned short* wkb = ws + off; off += nWk;
    unsigned short* wvb = ws + off; off += nWk;
    unsigned short* wob = ws + off; off += nWq;
    unsigned short* Qb  = ws + off; off += (size_t)TOKENS * QF;
    unsigned short* Kb  = ws + off; off += (size_t)TOKENS * KVF;
    unsigned short* Vt  = ws + off; off += (size_t)TOKENS * KVF;
    unsigned short* Ob  = ws + off; off += (size_t)TOKENS * QF;

    // 1) fp32 -> bf16 conversions
    auto cvt = [&](const float* s, unsigned short* d, size_t n) {
        int blocks = (int)((n + 255) / 256);
        cvt_f32_bf16<<<blocks, 256, 0, stream>>>(s, d, (int)n);
    };
    cvt(x,  xb,  nX);
    cvt(wq, wqb, nWq);
    cvt(wk, wkb, nWk);
    cvt(wv, wvb, nWk);
    cvt(wo, wob, nWq);

    // 2) projections: y = x @ W^T  (NT GEMM on bf16), 128x128 block tiles
    dim3 blk(128);
    gemm_nt_bf16<<<dim3(TOKENS / 128, QF / 128),  blk, 0, stream>>>(xb, wqb, Qb,
        TOKENS, QF, D_MODEL, 0);
    gemm_nt_bf16<<<dim3(TOKENS / 128, KVF / 128), blk, 0, stream>>>(xb, wkb, Kb,
        TOKENS, KVF, D_MODEL, 0);
    gemm_nt_bf16<<<dim3(TOKENS / 128, KVF / 128), blk, 0, stream>>>(xb, wvb, Vt,
        TOKENS, KVF, D_MODEL, 2);   // V stored transposed per head

    // 3) attention
    attn_gqa_swa<<<dim3(SEQ / 16, N_Q, BATCH), dim3(32), 0, stream>>>(Qb, Kb, Vt, Ob);

    // 4) output projection -> fp32 result
    gemm_nt_bf16<<<dim3(TOKENS / 128, QF / 128), blk, 0, stream>>>(Ob, wob, out,
        TOKENS, QF, D_MODEL, 1);
}
